// PIMACV3ActorRNN_41386304864870
// MI455X (gfx1250) — matrix operand order repaired and checked by reading, not compiled, not run on hardware
//
#include <hip/hip_runtime.h>

// ---------------------------------------------------------------------------
// MI455X (gfx1250) implementation of PIMACV3ActorRNN forward.
//
// Shapes: B=16, T=256 (4096 tokens), OBS=128, W0=256, HID=256, CTX=64, A=64,
// RANK=4, HN_H=128, HN_OUT=1344.
//
// bf16 WMMA (v_wmma_f32_16x16x32_bf16) with f32 accumulation everywhere.
// GRU scan: single 16-wave workgroup; each wave owns the r/z/n gate tiles of
// one 16-wide hidden slice, Whh register-stationary (24 fragments = 192
// VGPRs, fits the 256-VGPR budget at 4 waves/SIMD), fp32 hidden-state carry
// in registers, 8 KB LDS bf16 h-exchange, prefetch of next step's XG.
// ---------------------------------------------------------------------------

typedef __bf16 bf16_t;
typedef bf16_t v16bf __attribute__((ext_vector_type(16)));
typedef bf16_t v8bf  __attribute__((ext_vector_type(8)));
typedef float  v8f   __attribute__((ext_vector_type(8)));

#define NB   16
#define NT   256
#define NTOK 4096   // NB*NT
#define HID  256
#define G3   768    // 3*HID
#define HN_OUT 1344

__device__ __forceinline__ v8f wmma_bf16(v16bf a, v16bf b, v8f c) {
  // 8 args: (neg_a, A, neg_b, B, c_mod, C, reuse_a, reuse_b)
  return __builtin_amdgcn_wmma_f32_16x16x32_bf16(
      false, a, false, b, (short)0, c, false, false);
}

// A fragment (16x32 bf16, ISA 7.12.2): lane holds row M = lane%16;
// lanes 0-15: elems 0-15 = K 0..15, lanes 16-31: K 16..31 (contiguous order used here, matching frag_b).
__device__ __forceinline__ v16bf frag_a(const bf16_t* A, int K, int row,
                                        int kb, int lane) {
  const int half = (lane >> 4) & 1;
  const bf16_t* base = A + (size_t)row * K + kb * 32;
  v8bf lo = *(const v8bf*)(base + half * 16);
  v8bf hi = *(const v8bf*)(base + half * 16 + 8);
  v16bf r;
#pragma unroll
  for (int i = 0; i < 8; ++i) { r[i] = lo[i]; r[i + 8] = hi[i]; }
  return r;
}

// B fragment (32x16 bf16): lane holds column n = lane%16 (a row of W since
// we compute A @ W^T with W row-major [N][K]); half-wave selects K 0..15 vs
// K 16..31 (contiguous order used here, matching frag_a).
__device__ __forceinline__ v16bf frag_b(const bf16_t* W, int K, int col,
                                        int kb, int lane) {
  const int half = (lane >> 4) & 1;
  const bf16_t* base = W + (size_t)col * K + kb * 32 + half * 16;
  v8bf lo = *(const v8bf*)(base);
  v8bf hi = *(const v8bf*)(base + 8);
  v16bf r;
#pragma unroll
  for (int i = 0; i < 8; ++i) { r[i] = lo[i]; r[i + 8] = hi[i]; }
  return r;
}

// ---------------------------------------------------------------------------
// Pack fp32 -> bf16
// ---------------------------------------------------------------------------
__global__ void pack_bf16(const float* __restrict__ src,
                          bf16_t* __restrict__ dst, int n) {
  int i = blockIdx.x * blockDim.x + threadIdx.x;
  if (i < n) dst[i] = (bf16_t)src[i];
}

// ---------------------------------------------------------------------------
// Generic WMMA GEMM: C[M,N] = op(A[M,K] @ W[N,K]^T + bias[N])
// One wave per 16x64 output tile. OP: 0 relu->bf16, 1 ->bf16, 2 ->f32,
// 3 clip(-6,4)->f32, 4 *scale->f32.
// ---------------------------------------------------------------------------
template <int OP>
__global__ void gemm_wmma(const bf16_t* __restrict__ A,
                          const bf16_t* __restrict__ W,
                          const float* __restrict__ bias,
                          bf16_t* __restrict__ outb,
                          float* __restrict__ outf,
                          int N, int K, float scale) {
  const int lane  = threadIdx.x & 31;
  const int mtile = blockIdx.y;
  const int nbase = blockIdx.x * 64;
  const int row   = mtile * 16 + (lane & 15);

  v8f acc[4] = {v8f{}, v8f{}, v8f{}, v8f{}};
  const int nkb = K >> 5;
  for (int kb = 0; kb < nkb; ++kb) {
    v16bf a = frag_a(A, K, row, kb, lane);
#pragma unroll
    for (int t = 0; t < 4; ++t) {
      v16bf b = frag_b(W, K, nbase + t * 16 + (lane & 15), kb, lane);
      acc[t] = wmma_bf16(a, b, acc[t]);
    }
  }
  const int col0  = lane & 15;
  const int rbase = (lane >> 4) * 8;
#pragma unroll
  for (int t = 0; t < 4; ++t) {
    const int col = nbase + t * 16 + col0;
    const float bv = bias[col];
#pragma unroll
    for (int r = 0; r < 8; ++r) {
      const int m = mtile * 16 + rbase + r;
      float v = acc[t][r] + bv;
      if (OP == 0) v = fmaxf(v, 0.f);
      if (OP == 3) v = fminf(fmaxf(v, -6.f), 4.f);
      if (OP == 4) v = v * scale;
      if (OP == 0 || OP == 1)
        outb[(size_t)m * N + col] = (bf16_t)v;
      else
        outf[(size_t)m * N + col] = v;
    }
  }
}

// ---------------------------------------------------------------------------
// GRU scan. Single workgroup, 16 waves (512 threads). Wave w owns hidden
// columns [16w, 16w+16) and computes all three gate tiles (r, z, n) for that
// slice, so the nonlinearity runs entirely in registers on the WMMA
// accumulators (lane = hidden col, acc regs = batch rows). Whh is held
// register-stationary: 3 gates x 8 k-blocks = 24 v16bf fragments = 192 VGPRs,
// inside the 256-VGPR budget at 4 waves/SIMD. Hidden state: fp32 carry in
// registers, bf16 copy in LDS for cross-wave A-fragment exchange.
// ---------------------------------------------------------------------------
__global__ void __launch_bounds__(512, 1)
gru_scan(const bf16_t* __restrict__ Whh,   // [768,256] bf16
         const float*  __restrict__ XG,    // [NTOK,768]  x@Wih^T + bih
         const float*  __restrict__ bhh,   // [768]
         const float*  __restrict__ h0,    // [NB,HID]
         bf16_t* __restrict__ feats,       // [NTOK,HID] bf16
         float*  __restrict__ hT) {        // [NB,HID] -> d_out tail
  __shared__ bf16_t hb[NB * HID];          // 8 KB

  const int tid  = threadIdx.x;
  const int lane = tid & 31;
  const int wave = tid >> 5;               // 0..15 = hidden tile
  const int col0 = lane & 15;
  const int half = (lane >> 4) & 1;
  const int j    = wave * 16 + col0;       // owned hidden column

  // weight-stationary Whh fragments (r, z, n tiles of this hidden slice)
  v16bf br[8], bz[8], bn[8];
#pragma unroll
  for (int kb = 0; kb < 8; ++kb) {
    br[kb] = frag_b(Whh, HID, j,           kb, lane);
    bz[kb] = frag_b(Whh, HID, HID + j,     kb, lane);
    bn[kb] = frag_b(Whh, HID, 2 * HID + j, kb, lane);
  }
  const float bhr = bhh[j];
  const float bhz = bhh[HID + j];
  const float bhn = bhh[2 * HID + j];

  // fp32 hidden-state carry: this lane owns (b = r + half*8, j) for r=0..7
  float hreg[8];
#pragma unroll
  for (int r = 0; r < 8; ++r) {
    const int b = r + half * 8;
    hreg[r] = h0[b * HID + j];
    hb[b * HID + j] = (bf16_t)hreg[r];
  }
  __syncthreads();

  for (int t = 0; t < NT; ++t) {
    // --- GEMM: h @ Whh^T for this wave's three gate tiles ---
    v8f ar = v8f{}, az = v8f{}, an = v8f{};
#pragma unroll
    for (int kb = 0; kb < 8; ++kb) {
      v16bf a = frag_a(&hb[0], HID, col0, kb, lane);
      ar = wmma_bf16(a, br[kb], ar);
      az = wmma_bf16(a, bz[kb], az);
      an = wmma_bf16(a, bn[kb], an);
    }
    __syncthreads();  // all waves done reading hb

    // --- gate math in registers; D-layout: reg r <-> batch b = r + half*8 ---
#pragma unroll
    for (int r = 0; r < 8; ++r) {
      const int b = r + half * 8;
      const size_t tok = (size_t)b * NT + t;
      const float xr = XG[tok * G3 + j];
      const float xz = XG[tok * G3 + HID + j];
      const float xn = XG[tok * G3 + 2 * HID + j];
      if (t + 1 < NT) {  // overlap next step's XG fetch with the next step's WMMA chain
        __builtin_prefetch(&XG[(tok + 1) * G3 + j], 0, 1);
        __builtin_prefetch(&XG[(tok + 1) * G3 + HID + j], 0, 1);
        __builtin_prefetch(&XG[(tok + 1) * G3 + 2 * HID + j], 0, 1);
      }
      const float r_ = 1.f / (1.f + expf(-(xr + ar[r] + bhr)));
      const float z_ = 1.f / (1.f + expf(-(xz + az[r] + bhz)));
      const float n_ = tanhf(xn + r_ * (an[r] + bhn));
      const float hnew = (1.f - z_) * n_ + z_ * hreg[r];
      hreg[r] = hnew;
      hb[b * HID + j] = (bf16_t)hnew;
      feats[tok * HID + j] = (bf16_t)hnew;
      if (t == NT - 1) hT[b * HID + j] = hnew;
    }
    __syncthreads();
  }
}

// ---------------------------------------------------------------------------
// Finalize: per token (block of 64 threads):
//   gate = sigmoid(-gate_w * mean(ctx_lv) + gate_b)
//   s[r] = feats . wf[:,r]           (rank-4 contraction, LDS reduce)
//   logits[a] = P[a] + gate*(sum_r s[r]*af[r,a] + db[a])
// (P already contains feats@W_pol^T + b_pol)
// ---------------------------------------------------------------------------
__global__ void finalize(const bf16_t* __restrict__ feats,
                         const float* __restrict__ LV,    // [NTOK,64] clipped
                         const float* __restrict__ D,     // [NTOK,1344] deltas
                         const float* __restrict__ P,     // [NTOK,64]
                         const float* __restrict__ gw,
                         const float* __restrict__ gb,
                         float* __restrict__ logits) {
  const int t = blockIdx.x;
  const int a = threadIdx.x;   // 0..63
  __shared__ float red[64];
  __shared__ float s[4];
  __shared__ float gsh;

  // gate
  red[a] = LV[(size_t)t * 64 + a];
  __syncthreads();
  for (int off = 32; off > 0; off >>= 1) {
    if (a < off) red[a] += red[a + off];
    __syncthreads();
  }
  if (a == 0) {
    const float mean = red[0] * (1.f / 64.f);
    gsh = 1.f / (1.f + expf(-((*gw) * (-mean) + (*gb))));
  }

  // s[r] = sum_h feats[h] * wf[h][r]
  float sp[4] = {0.f, 0.f, 0.f, 0.f};
#pragma unroll
  for (int q = 0; q < 4; ++q) {
    const int h = a + q * 64;
    const float f = (float)feats[(size_t)t * HID + h];
    const float* wf = D + (size_t)t * HN_OUT + (size_t)h * 4;
    sp[0] += f * wf[0]; sp[1] += f * wf[1];
    sp[2] += f * wf[2]; sp[3] += f * wf[3];
  }
  __syncthreads();
#pragma unroll
  for (int r = 0; r < 4; ++r) {
    red[a] = sp[r];
    __syncthreads();
    for (int off = 32; off > 0; off >>= 1) {
      if (a < off) red[a] += red[a + off];
      __syncthreads();
    }
    if (a == 0) s[r] = red[0];
    __syncthreads();
  }

  const float gate = gsh;
  const float* af = D + (size_t)t * HN_OUT + 1024;
  const float da = s[0] * af[0 * 64 + a] + s[1] * af[1 * 64 + a] +
                   s[2] * af[2 * 64 + a] + s[3] * af[3 * 64 + a];
  const float db = D[(size_t)t * HN_OUT + 1280 + a];
  logits[(size_t)t * 64 + a] = P[(size_t)t * 64 + a] + gate * (da + db);
}

// ---------------------------------------------------------------------------
// Host launch
// ---------------------------------------------------------------------------
static inline size_t wsalloc(size_t& off, size_t bytes) {
  size_t o = off;
  off += (bytes + 255) & ~(size_t)255;
  return o;
}

extern "C" void kernel_launch(void* const* d_in, const int* in_sizes, int n_in,
                              void* d_out, int out_size, void* d_ws,
                              size_t ws_size, hipStream_t stream) {
  (void)in_sizes; (void)n_in; (void)out_size; (void)ws_size;
  const float* obs   = (const float*)d_in[0];
  const float* h0    = (const float*)d_in[1];
  const float* W_in  = (const float*)d_in[2];
  const float* b_in  = (const float*)d_in[3];
  const float* W_h1  = (const float*)d_in[4];
  const float* b_h1  = (const float*)d_in[5];
  const float* Wih   = (const float*)d_in[6];
  const float* Whh   = (const float*)d_in[7];
  const float* bih   = (const float*)d_in[8];
  const float* bhh   = (const float*)d_in[9];
  const float* W_mu  = (const float*)d_in[10];
  const float* b_mu  = (const float*)d_in[11];
  const float* W_lv  = (const float*)d_in[12];
  const float* b_lv  = (const float*)d_in[13];
  const float* gw    = (const float*)d_in[14];
  const float* gb    = (const float*)d_in[15];
  const float* W_pol = (const float*)d_in[16];
  const float* b_pol = (const float*)d_in[17];
  const float* hn_W1 = (const float*)d_in[18];
  const float* hn_b1 = (const float*)d_in[19];
  const float* hn_W2 = (const float*)d_in[20];
  const float* hn_b2 = (const float*)d_in[21];
  const float* hn_W3 = (const float*)d_in[22];
  const float* hn_b3 = (const float*)d_in[23];

  float* out = (float*)d_out;                 // logits [4096,64] then hT [16,256]
  char*  ws  = (char*)d_ws;
  size_t off = 0;

  bf16_t* obs_b  = (bf16_t*)(ws + wsalloc(off, (size_t)NTOK * 128 * 2));
  bf16_t* Win_b  = (bf16_t*)(ws + wsalloc(off, (size_t)256 * 128 * 2));
  bf16_t* Wh1_b  = (bf16_t*)(ws + wsalloc(off, (size_t)256 * 256 * 2));
  bf16_t* Wih_b  = (bf16_t*)(ws + wsalloc(off, (size_t)G3 * 256 * 2));
  bf16_t* Whh_b  = (bf16_t*)(ws + wsalloc(off, (size_t)G3 * 256 * 2));
  bf16_t* Wmu_b  = (bf16_t*)(ws + wsalloc(off, (size_t)64 * 256 * 2));
  bf16_t* Wlv_b  = (bf16_t*)(ws + wsalloc(off, (size_t)64 * 256 * 2));
  bf16_t* Wpol_b = (bf16_t*)(ws + wsalloc(off, (size_t)64 * 256 * 2));
  bf16_t* W1_b   = (bf16_t*)(ws + wsalloc(off, (size_t)128 * 64 * 2));
  bf16_t* W2_b   = (bf16_t*)(ws + wsalloc(off, (size_t)128 * 128 * 2));
  bf16_t* W3_b   = (bf16_t*)(ws + wsalloc(off, (size_t)HN_OUT * 128 * 2));
  bf16_t* X1     = (bf16_t*)(ws + wsalloc(off, (size_t)NTOK * 256 * 2));
  bf16_t* X2     = (bf16_t*)(ws + wsalloc(off, (size_t)NTOK * 256 * 2));
  float*  XG     = (float*) (ws + wsalloc(off, (size_t)NTOK * G3 * 4));
  bf16_t* FE     = (bf16_t*)(ws + wsalloc(off, (size_t)NTOK * HID * 2));
  bf16_t* MU     = (bf16_t*)(ws + wsalloc(off, (size_t)NTOK * 64 * 2));
  float*  LV     = (float*) (ws + wsalloc(off, (size_t)NTOK * 64 * 4));
  bf16_t* H1     = (bf16_t*)(ws + wsalloc(off, (size_t)NTOK * 128 * 2));
  bf16_t* H2     = (bf16_t*)(ws + wsalloc(off, (size_t)NTOK * 128 * 2));
  float*  DL     = (float*) (ws + wsalloc(off, (size_t)NTOK * HN_OUT * 4));
  float*  PO     = (float*) (ws + wsalloc(off, (size_t)NTOK * 64 * 4));

  auto pack = [&](const float* s, bf16_t* d, int n) {
    pack_bf16<<<(n + 255) / 256, 256, 0, stream>>>(s, d, n);
  };
  pack(obs,   obs_b,  NTOK * 128);
  pack(W_in,  Win_b,  256 * 128);
  pack(W_h1,  Wh1_b,  256 * 256);
  pack(Wih,   Wih_b,  G3 * 256);
  pack(Whh,   Whh_b,  G3 * 256);
  pack(W_mu,  Wmu_b,  64 * 256);
  pack(W_lv,  Wlv_b,  64 * 256);
  pack(W_pol, Wpol_b, 64 * 256);
  pack(hn_W1, W1_b,   128 * 64);
  pack(hn_W2, W2_b,   128 * 128);
  pack(hn_W3, W3_b,   HN_OUT * 128);

  const dim3 blk(32);
  const int MT = NTOK / 16;  // 256 m-tiles
  // X1 = relu(obs @ W_in^T + b_in)            [4096,256], K=128
  gemm_wmma<0><<<dim3(256 / 64, MT), blk, 0, stream>>>(obs_b, Win_b, b_in, X1, nullptr, 256, 128, 1.f);
  // X2 = relu(X1 @ W_h1^T + b_h1)             [4096,256], K=256
  gemm_wmma<0><<<dim3(256 / 64, MT), blk, 0, stream>>>(X1, Wh1_b, b_h1, X2, nullptr, 256, 256, 1.f);
  // XG = X2 @ Wih^T + bih                     [4096,768], K=256, f32
  gemm_wmma<2><<<dim3(G3 / 64, MT), blk, 0, stream>>>(X2, Wih_b, bih, nullptr, XG, G3, 256, 1.f);
  // GRU scan -> feats (bf16) and hT (fp32 tail of d_out)
  gru_scan<<<1, 512, 0, stream>>>(Whh_b, XG, bhh, h0, FE, out + (size_t)NTOK * 64);
  // ctx_mu = feats @ W_mu^T + b_mu            [4096,64] bf16
  gemm_wmma<1><<<dim3(1, MT), blk, 0, stream>>>(FE, Wmu_b, b_mu, MU, nullptr, 64, 256, 1.f);
  // ctx_lv = clip(feats @ W_lv^T + b_lv)      [4096,64] f32
  gemm_wmma<3><<<dim3(1, MT), blk, 0, stream>>>(FE, Wlv_b, b_lv, nullptr, LV, 64, 256, 1.f);
  // h1 = relu(ctx_mu @ hn_W1^T + hn_b1)       [4096,128], K=64
  gemm_wmma<0><<<dim3(2, MT), blk, 0, stream>>>(MU, W1_b, hn_b1, H1, nullptr, 128, 64, 1.f);
  // h2 = relu(h1 @ hn_W2^T + hn_b2)           [4096,128], K=128
  gemm_wmma<0><<<dim3(2, MT), blk, 0, stream>>>(H1, W2_b, hn_b2, H2, nullptr, 128, 128, 1.f);
  // deltas = (h2 @ hn_W3^T + hn_b3) * 0.05    [4096,1344], K=128, f32
  gemm_wmma<4><<<dim3(HN_OUT / 64, MT), blk, 0, stream>>>(H2, W3_b, hn_b3, nullptr, DL, HN_OUT, 128, 0.05f);
  // P = feats @ W_pol^T + b_pol               [4096,64], f32
  gemm_wmma<2><<<dim3(1, MT), blk, 0, stream>>>(FE, Wpol_b, b_pol, nullptr, PO, 64, 256, 1.f);
  // logits
  finalize<<<NTOK, 64, 0, stream>>>(FE, LV, DL, PO, gw, gb, out);
}
